// Skeleton_Encoder_27410481283594
// MI455X (gfx1250) — matrix operand order repaired and checked by reading, not compiled, not run on hardware
//
#include <hip/hip_runtime.h>
#include <hip/hip_bf16.h>
#include <math.h>

typedef __attribute__((ext_vector_type(16))) _Float16 v16h;
typedef __attribute__((ext_vector_type(8)))  float    v8f;

#define B_  64
#define C_  3
#define T_  256
#define V_  21
#define D_  128
#define H_  4
#define KT_ 9
#define VP_ 32
#define TT_ 32
#define NFRM (B_ * T_)                       // 16384
#define ACT_ELEMS ((size_t)NFRM * V_ * D_)   // 44,040,192
#define FRAG_ (32 * 16)                      // f16 elems per packed 32x16 B fragment

// ---------------- WMMA helpers (wave32, v_wmma_f32_16x16x32_f16) ----------------

__device__ __forceinline__ v8f wmma16(v16h a, v16h b, v8f c) {
  return __builtin_amdgcn_wmma_f32_16x16x32_f16(false, a, false, b, (short)0, c,
                                                false, false);
}

// A fragment: 16x32 f16 tile, row-major source base[m*lda + k]
// (per lane: one contiguous 16-f16 run, k = 16*h + s)
__device__ __forceinline__ v16h load_a(const _Float16* base, int lda) {
  int lane = threadIdx.x & 31;
  int m = lane & 15, h = lane >> 4;
  const _Float16* row = base + m * lda;
  v16h a;
#pragma unroll
  for (int j = 0; j < 8; ++j) {
    int k0 = h * 16 + 2 * j;
    a[2 * j]     = row[k0];
    a[2 * j + 1] = row[k0 + 1];
  }
  return a;
}

// Packed B fragment: one contiguous 32-byte load per lane.
// Layout: base[frag*512 + lane*16 + e] = B[k = 32*ks + (lane>>4)*16 + e][n = (lane&15)]
__device__ __forceinline__ v16h load_bp(const _Float16* base, int frag) {
  return *(const v16h*)(base + ((size_t)frag * 32 + (threadIdx.x & 31)) * 16);
}

// ---------------- weight prep: fp32 -> f16 fragment-packed B layouts ----------------

// 128x128 weight -> packed [ks(4)][nt(8)][lane(32)][e(16)]
// transpose=0: B[k][n] = w[k*128+n];  transpose=1: B[k][n] = w[n*128+k]
__global__ void k_pack_w(const float* __restrict__ w, _Float16* __restrict__ d,
                         int transpose) {
  int i = blockIdx.x * 256 + threadIdx.x;
  if (i >= 4 * 8 * FRAG_) return;
  int e = i & 15, lane = (i >> 4) & 31, nt = (i >> 9) & 7, ks = i >> 12;
  int k = ks * 32 + ((lane >> 4) << 4) + e;
  int n = nt * 16 + (lane & 15);
  d[i] = (_Float16)(transpose ? w[n * D_ + k] : w[k * D_ + n]);
}

// tcn im2col weight: B[kk = ktap*128+c][n] = tw[(n*128+c)*9 + ktap]
// packed [ks(36)][nt(8)][lane][e]
__global__ void k_pack_tcn(const float* __restrict__ tw, _Float16* __restrict__ d) {
  int i = blockIdx.x * 256 + threadIdx.x;
  if (i >= 36 * 8 * FRAG_) return;
  int e = i & 15, lane = (i >> 4) & 31, nt = (i >> 9) & 7, ks = i >> 12;
  int kk = ks * 32 + ((lane >> 4) << 4) + e;
  int n = nt * 16 + (lane & 15);
  int ktap = kk >> 7, c = kk & 127;
  d[i] = (_Float16)tw[(n * D_ + c) * KT_ + ktap];
}

// padded transposed adjacency (row-major, used as A tile): d[w*32+v] = A[v*21+w]
__global__ void k_adj(const float* __restrict__ A, _Float16* __restrict__ d) {
  int i = blockIdx.x * 256 + threadIdx.x;
  if (i < VP_ * VP_) {
    int w = i >> 5, v = i & 31;
    d[i] = (_Float16)((w < V_ && v < V_) ? A[v * V_ + w] : 0.f);
  }
}

// ---------------- encoder: joint embed + MHA + output proj ----------------

__global__ __launch_bounds__(128) void k_encoder(
    const float* __restrict__ x, const float* __restrict__ coord_w,
    const float* __restrict__ coord_b, const float* __restrict__ joint_emb,
    const float* __restrict__ graph_bias,
    const float* __restrict__ bq, const float* __restrict__ bk,
    const float* __restrict__ bv, const float* __restrict__ bo,
    const _Float16* __restrict__ WQ, const _Float16* __restrict__ WK,
    const _Float16* __restrict__ WV, const _Float16* __restrict__ WO,
    _Float16* __restrict__ act_out, float* __restrict__ attn_out)
{
  __shared__ float    sCo[C_][V_];           // raw coords
  __shared__ _Float16 sF[VP_][D_];           // feat (row-major A), reused for attn-out
  __shared__ _Float16 sQ[VP_][D_];           // Q row-major (A for scores)
  __shared__ _Float16 sKt[8 * FRAG_];        // K^T packed-B: frag = head*2 + jtile
  __shared__ _Float16 sVp[8 * FRAG_];        // V packed-B: frag = column-tile (0..7)
  __shared__ float    sS[H_][VP_][VP_];      // scores
  __shared__ _Float16 sP[H_][VP_][VP_];      // probs row-major (A for attn@V)

  const int n = blockIdx.x;
  const int b = n / T_, t = n % T_;
  const int tid = threadIdx.x;
  const int lane = tid & 31;
  const int wid = tid >> 5;        // wave 0..3 == head id in attention stages
  const int ln = lane & 15;        // C-frag column
  const int lh = lane >> 4;        // C-frag half

  if (tid < C_ * V_) {
    int c = tid / V_, v = tid % V_;
    sCo[c][v] = x[((b * C_ + c) * T_ + t) * V_ + v];
  }
  __syncthreads();

  // 1. feat = coords @ coord_w + coord_b + joint_emb (K=3, scalar)
  for (int idx = tid; idx < VP_ * D_; idx += 128) {
    int v = idx >> 7, d = idx & 127;
    float f = 0.f;
    if (v < V_) {
      f = sCo[0][v] * coord_w[d] + sCo[1][v] * coord_w[D_ + d]
        + sCo[2][v] * coord_w[2 * D_ + d] + coord_b[d] + joint_emb[v * D_ + d];
    }
    sF[v][d] = (_Float16)f;
  }
  __syncthreads();

  // 2. Q/K/V projections, wave wid owns columns [wid*32, wid*32+32)
  {
    const _Float16* Wm[3] = {WQ, WK, WV};
    const float*    bm[3] = {bq, bk, bv};
#pragma unroll
    for (int mat = 0; mat < 3; ++mat) {
      v8f acc[2][2] = {};
#pragma unroll
      for (int ks = 0; ks < 4; ++ks) {
        v16h a0 = load_a(&sF[0][ks * 32], D_);
        v16h a1 = load_a(&sF[16][ks * 32], D_);
        v16h b0 = load_bp(Wm[mat], ks * 8 + wid * 2 + 0);
        v16h b1 = load_bp(Wm[mat], ks * 8 + wid * 2 + 1);
        acc[0][0] = wmma16(a0, b0, acc[0][0]);
        acc[0][1] = wmma16(a0, b1, acc[0][1]);
        acc[1][0] = wmma16(a1, b0, acc[1][0]);
        acc[1][1] = wmma16(a1, b1, acc[1][1]);
      }
#pragma unroll
      for (int mt = 0; mt < 2; ++mt) {
#pragma unroll
        for (int nt = 0; nt < 2; ++nt) {
          float bias = bm[mat][wid * 32 + nt * 16 + ln];
#pragma unroll
          for (int r = 0; r < 8; ++r) {
            int m = mt * 16 + r + 8 * lh;            // row of Q/K/V
            int nc = wid * 32 + nt * 16 + ln;        // global column
            _Float16 val = (_Float16)(acc[mt][nt][r] + bias);
            if (mat == 0) {
              sQ[m][nc] = val;
            } else if (mat == 1) {
              // K^T packed-B: k = dh_local, n' = joint m
              int dl = nt * 16 + ln;                 // dh_local 0..31 (head == wid)
              sKt[(((wid * 2 + (m >> 4)) * 32) + ((dl >> 4) << 4) + (m & 15)) * 16
                  + (dl & 15)] = val;
            } else {
              // V packed-B: k = joint m, column tile = nc>>4
              sVp[(((nc >> 4) * 32) + ((m >> 4) << 4) + (nc & 15)) * 16
                  + (m & 15)] = val;
            }
          }
        }
      }
    }
  }
  __syncthreads();

  // 3. scores[h=wid] = (Q_h @ K_h^T) * scale + graph_bias
  {
    const float scale = 0.1767766952966369f;  // 1/sqrt(32)
#pragma unroll
    for (int mt = 0; mt < 2; ++mt) {
#pragma unroll
      for (int nt = 0; nt < 2; ++nt) {
        v8f acc = {};
        v16h a  = load_a(&sQ[mt * 16][wid * 32], D_);
        v16h bf = load_bp(sKt, wid * 2 + nt);
        acc = wmma16(a, bf, acc);
        int j = nt * 16 + ln;
#pragma unroll
        for (int r = 0; r < 8; ++r) {
          int i = mt * 16 + r + 8 * lh;
          float gbv = (i < V_ && j < V_) ? graph_bias[i * V_ + j] : 0.f;
          sS[wid][i][j] = acc[r] * scale + gbv;
        }
      }
    }
  }
  {  // zero prob tiles so padded rows/cols contribute nothing
    _Float16* p = &sP[0][0][0];
    for (int idx = tid; idx < H_ * VP_ * VP_; idx += 128) p[idx] = (_Float16)0.f;
  }
  __syncthreads();

  // 4. softmax over j<21 (one thread per (h,i) row); attn straight to d_out
  if (tid < H_ * V_) {
    int h = tid / V_, i = tid % V_;
    float mx = -1e30f;
    for (int j = 0; j < V_; ++j) mx = fmaxf(mx, sS[h][i][j]);
    float sum = 0.f;
    for (int j = 0; j < V_; ++j) {
      float e = __expf(sS[h][i][j] - mx);
      sS[h][i][j] = e;
      sum += e;
    }
    float inv = 1.f / sum;
    size_t abase = (((size_t)n * H_ + h) * V_ + i) * V_;
    for (int j = 0; j < V_; ++j) {
      float pv = sS[h][i][j] * inv;
      sP[h][i][j] = (_Float16)pv;
      attn_out[abase + j] = pv;
    }
  }
  __syncthreads();

  // 5. out_h = P_h @ V_h (K = 32 padded joints) -> reuse sF (row-major)
#pragma unroll
  for (int mt = 0; mt < 2; ++mt) {
#pragma unroll
    for (int nt = 0; nt < 2; ++nt) {
      v8f acc = {};
      v16h a  = load_a(&sP[wid][mt * 16][0], VP_);
      v16h bf = load_bp(sVp, wid * 2 + nt);
      acc = wmma16(a, bf, acc);
#pragma unroll
      for (int r = 0; r < 8; ++r)
        sF[mt * 16 + r + 8 * lh][wid * 32 + nt * 16 + ln] = (_Float16)acc[r];
    }
  }
  __syncthreads();

  // 6. feat = out @ wo + bo -> f16 activation buffer [B,T,V,D]
  {
    v8f acc[2][2] = {};
#pragma unroll
    for (int ks = 0; ks < 4; ++ks) {
      v16h a0 = load_a(&sF[0][ks * 32], D_);
      v16h a1 = load_a(&sF[16][ks * 32], D_);
      v16h b0 = load_bp(WO, ks * 8 + wid * 2 + 0);
      v16h b1 = load_bp(WO, ks * 8 + wid * 2 + 1);
      acc[0][0] = wmma16(a0, b0, acc[0][0]);
      acc[0][1] = wmma16(a0, b1, acc[0][1]);
      acc[1][0] = wmma16(a1, b0, acc[1][0]);
      acc[1][1] = wmma16(a1, b1, acc[1][1]);
    }
#pragma unroll
    for (int mt = 0; mt < 2; ++mt) {
#pragma unroll
      for (int nt = 0; nt < 2; ++nt) {
        const int n0 = wid * 32 + nt * 16;
        float bias = bo[n0 + ln];
#pragma unroll
        for (int r = 0; r < 8; ++r) {
          int v = mt * 16 + r + 8 * lh;
          if (v < V_)
            act_out[((size_t)n * V_ + v) * D_ + n0 + ln] =
                (_Float16)(acc[mt][nt][r] + bias);
        }
      }
    }
  }
}

// ---------------- spatial graph conv: relu((A^T @ X) @ gw^T + gb) ----------------

__global__ __launch_bounds__(128) void k_gcn(
    const _Float16* __restrict__ act_in, const _Float16* __restrict__ adjT,
    const _Float16* __restrict__ GW, const float* __restrict__ gb,
    _Float16* __restrict__ act_out)
{
  __shared__ _Float16 sXp[8 * FRAG_];        // X in packed-B layout (frag = col tile)
  __shared__ _Float16 sAgg[VP_][D_];         // aggregate, row-major (A for stage 2)
  __shared__ _Float16 sAdj[VP_][VP_];

  const int n = blockIdx.x;
  const int tid = threadIdx.x, lane = tid & 31, wid = tid >> 5;
  const int ln = lane & 15, lh = lane >> 4;

  // load X straight into packed-B layout: elem e of lane holds X[v=(lane>>4)*16+e][d]
  for (int idx = tid; idx < 8 * FRAG_; idx += 128) {
    int e = idx & 15, l = (idx >> 4) & 31, ntg = idx >> 9;
    int v = ((l >> 4) << 4) + e;
    int d = ntg * 16 + (l & 15);
    sXp[idx] = (v < V_) ? act_in[((size_t)n * V_ + v) * D_ + d] : (_Float16)0.f;
  }
  for (int idx = tid; idx < VP_ * VP_; idx += 128)
    (&sAdj[0][0])[idx] = adjT[idx];
  __syncthreads();

  // stage 1: agg = A^T @ X (M=32, K=32, N=128)
#pragma unroll
  for (int mt = 0; mt < 2; ++mt) {
#pragma unroll
    for (int nt = 0; nt < 2; ++nt) {
      v8f acc = {};
      v16h a  = load_a(&sAdj[mt * 16][0], VP_);
      v16h bf = load_bp(sXp, wid * 2 + nt);
      acc = wmma16(a, bf, acc);
#pragma unroll
      for (int r = 0; r < 8; ++r)
        sAgg[mt * 16 + r + 8 * lh][wid * 32 + nt * 16 + ln] = (_Float16)acc[r];
    }
  }
  __syncthreads();

  // stage 2: out = relu(agg @ gw^T + gb)
  {
    v8f acc[2][2] = {};
#pragma unroll
    for (int ks = 0; ks < 4; ++ks) {
      v16h a0 = load_a(&sAgg[0][ks * 32], D_);
      v16h a1 = load_a(&sAgg[16][ks * 32], D_);
      v16h b0 = load_bp(GW, ks * 8 + wid * 2 + 0);
      v16h b1 = load_bp(GW, ks * 8 + wid * 2 + 1);
      acc[0][0] = wmma16(a0, b0, acc[0][0]);
      acc[0][1] = wmma16(a0, b1, acc[0][1]);
      acc[1][0] = wmma16(a1, b0, acc[1][0]);
      acc[1][1] = wmma16(a1, b1, acc[1][1]);
    }
#pragma unroll
    for (int mt = 0; mt < 2; ++mt) {
#pragma unroll
      for (int nt = 0; nt < 2; ++nt) {
        const int n0 = wid * 32 + nt * 16;
        float bias = gb[n0 + ln];
#pragma unroll
        for (int r = 0; r < 8; ++r) {
          int v = mt * 16 + r + 8 * lh;
          if (v < V_)
            act_out[((size_t)n * V_ + v) * D_ + n0 + ln] =
                (_Float16)fmaxf(acc[mt][nt][r] + bias, 0.f);
        }
      }
    }
  }
}

// ------------- temporal conv (im2col GEMM, K = 9*128) + bias + residual + relu -------------

__global__ __launch_bounds__(128) void k_tcn(
    const _Float16* __restrict__ act_in, const _Float16* __restrict__ res_in,
    const _Float16* __restrict__ TB, const float* __restrict__ tb,
    _Float16* __restrict__ act_out, float* __restrict__ final_out, int write_final)
{
  __shared__ _Float16 sY[TT_ + KT_ - 1][D_];   // 40 x 128 halo tile

  const int blk  = blockIdx.x;
  const int tblk = blk & 7;             // T_/TT_ = 8
  const int v    = (blk >> 3) % V_;
  const int b    = blk / (8 * V_);
  const int t0   = tblk * TT_;
  const int tid = threadIdx.x, lane = tid & 31, wid = tid >> 5;
  const int ln = lane & 15, lh = lane >> 4;

  for (int idx = tid; idx < (TT_ + KT_ - 1) * D_; idx += 128) {
    int r = idx >> 7, d = idx & 127;
    int tt = t0 - (KT_ / 2) + r;
    sY[r][d] = (tt >= 0 && tt < T_)
                   ? act_in[(((size_t)b * T_ + tt) * V_ + v) * D_ + d]
                   : (_Float16)0.f;
  }
  __syncthreads();

  v8f acc[2][2] = {};
  for (int ks = 0; ks < 36; ++ks) {           // K = 1152
    int k  = ks >> 2;                         // temporal tap 0..8
    int c0 = (ks & 3) * 32;                   // channel sub-block
    v16h a0 = load_a(&sY[k][c0], D_);
    v16h a1 = load_a(&sY[16 + k][c0], D_);
    v16h b0 = load_bp(TB, ks * 8 + wid * 2 + 0);
    v16h b1 = load_bp(TB, ks * 8 + wid * 2 + 1);
    acc[0][0] = wmma16(a0, b0, acc[0][0]);
    acc[0][1] = wmma16(a0, b1, acc[0][1]);
    acc[1][0] = wmma16(a1, b0, acc[1][0]);
    acc[1][1] = wmma16(a1, b1, acc[1][1]);
  }

#pragma unroll
  for (int mt = 0; mt < 2; ++mt) {
#pragma unroll
    for (int nt = 0; nt < 2; ++nt) {
      const int col = wid * 32 + nt * 16 + ln;
      float bias = tb[col];
#pragma unroll
      for (int r = 0; r < 8; ++r) {
        int t = t0 + mt * 16 + r + 8 * lh;
        size_t ridx = (((size_t)b * T_ + t) * V_ + v) * D_ + col;
        float val = fmaxf(acc[mt][nt][r] + bias + (float)res_in[ridx], 0.f);
        if (write_final)
          final_out[(((size_t)b * D_ + col) * T_ + t) * V_ + v] = val;   // [B,D,T,V]
        else
          act_out[ridx] = (_Float16)val;
      }
    }
  }
}

// ---------------- launch ----------------

extern "C" void kernel_launch(void* const* d_in, const int* in_sizes, int n_in,
                              void* d_out, int out_size, void* d_ws, size_t ws_size,
                              hipStream_t stream)
{
  (void)in_sizes; (void)n_in; (void)out_size; (void)ws_size;

  const float* x          = (const float*)d_in[0];
  const float* coord_w    = (const float*)d_in[1];
  const float* coord_b    = (const float*)d_in[2];
  const float* joint_emb  = (const float*)d_in[3];
  const float* graph_bias = (const float*)d_in[4];
  const float* wq  = (const float*)d_in[5];  const float* bq  = (const float*)d_in[6];
  const float* wk  = (const float*)d_in[7];  const float* bk  = (const float*)d_in[8];
  const float* wv  = (const float*)d_in[9];  const float* bv  = (const float*)d_in[10];
  const float* wo  = (const float*)d_in[11]; const float* bo  = (const float*)d_in[12];
  const float* gw1 = (const float*)d_in[13]; const float* gb1 = (const float*)d_in[14];
  const float* tw1 = (const float*)d_in[15]; const float* tb1 = (const float*)d_in[16];
  const float* gw2 = (const float*)d_in[17]; const float* gb2 = (const float*)d_in[18];
  const float* tw2 = (const float*)d_in[19]; const float* tb2 = (const float*)d_in[20];
  const float* Adj = (const float*)d_in[21];

  // fragment-packed f16 weight area at front of workspace
  _Float16* WQp = (_Float16*)d_ws;            // 4*8*512 = 16384
  _Float16* WKp = WQp + D_ * D_;
  _Float16* WVp = WKp + D_ * D_;
  _Float16* WOp = WVp + D_ * D_;
  _Float16* G1p = WOp + D_ * D_;
  _Float16* G2p = G1p + D_ * D_;
  _Float16* T1p = G2p + D_ * D_;              // 36*8*512 = 147456
  _Float16* T2p = T1p + KT_ * D_ * D_;
  _Float16* ADJp = T2p + KT_ * D_ * D_;       // 1024
  size_t wbytes = ((size_t)(6 * D_ * D_ + 2 * KT_ * D_ * D_ + VP_ * VP_)) * 2;
  _Float16* act0 = (_Float16*)((char*)d_ws + ((wbytes + 255) / 256) * 256);
  _Float16* act1 = act0 + ACT_ELEMS;
  _Float16* act2 = act1 + ACT_ELEMS;

  float* feat_out = (float*)d_out;
  float* attn_out = feat_out + ACT_ELEMS;

  // ---- weight prep (fragment packing) ----
  k_pack_w<<<(D_ * D_ + 255) / 256, 256, 0, stream>>>(wq, WQp, 0);
  k_pack_w<<<(D_ * D_ + 255) / 256, 256, 0, stream>>>(wk, WKp, 0);
  k_pack_w<<<(D_ * D_ + 255) / 256, 256, 0, stream>>>(wv, WVp, 0);
  k_pack_w<<<(D_ * D_ + 255) / 256, 256, 0, stream>>>(wo, WOp, 0);
  k_pack_w<<<(D_ * D_ + 255) / 256, 256, 0, stream>>>(gw1, G1p, 1);
  k_pack_w<<<(D_ * D_ + 255) / 256, 256, 0, stream>>>(gw2, G2p, 1);
  k_pack_tcn<<<(KT_ * D_ * D_ + 255) / 256, 256, 0, stream>>>(tw1, T1p);
  k_pack_tcn<<<(KT_ * D_ * D_ + 255) / 256, 256, 0, stream>>>(tw2, T2p);
  k_adj<<<(VP_ * VP_ + 255) / 256, 256, 0, stream>>>(Adj, ADJp);

  // ---- encoder ----
  k_encoder<<<NFRM, 128, 0, stream>>>(x, coord_w, coord_b, joint_emb, graph_bias,
                                      bq, bk, bv, bo, WQp, WKp, WVp, WOp,
                                      act0, attn_out);

  // ---- ST-GCN block 1 ----
  k_gcn<<<NFRM, 128, 0, stream>>>(act0, ADJp, G1p, gb1, act1);
  k_tcn<<<B_ * V_ * 8, 128, 0, stream>>>(act1, act0, T1p, tb1, act2, feat_out, 0);

  // ---- ST-GCN block 2 (final pass writes fp32 [B,D,T,V] to d_out) ----
  k_gcn<<<NFRM, 128, 0, stream>>>(act2, ADJp, G2p, gb2, act0);
  k_tcn<<<B_ * V_ * 8, 128, 0, stream>>>(act0, act2, T2p, tb2, act1, feat_out, 1);
}
